// Attention_18648747999429
// MI455X (gfx1250) — matrix operand order repaired and checked by reading, not compiled, not run on hardware
//
#include <hip/hip_runtime.h>
#include <hip/hip_bf16.h>
#include <stdint.h>

#define HID   512
#define BATCH 64
#define SRC   2048
#define MTILE 64

typedef __attribute__((ext_vector_type(16))) __bf16 v16b;
typedef __attribute__((ext_vector_type(8)))  float  v8f;

union ABFrag { uint4 q[2]; v16b v; };

__device__ __forceinline__ unsigned short f2bf(float f) {
  union { float f; unsigned u; } x; x.f = f;
  unsigned u = x.u;
  return (unsigned short)((u + 0x7FFFu + ((u >> 16) & 1u)) >> 16);  // RNE truncate
}
__device__ __forceinline__ unsigned pack2bf(float a, float b) {
  return (unsigned)f2bf(a) | ((unsigned)f2bf(b) << 16);
}
__device__ __forceinline__ float fast_tanh(float x) {
#if __has_builtin(__builtin_amdgcn_tanhf)
  return __builtin_amdgcn_tanhf(x);   // gfx1250 v_tanh_f32
#else
  return tanhf(x);
#endif
}

// ---- CDNA5 async global->LDS copy (ASYNCcnt) --------------------------------
typedef int vi4 __attribute__((vector_size(16)));
typedef __attribute__((address_space(1))) vi4* gv4p;   // int4 in global mem
typedef __attribute__((address_space(3))) vi4* lv4p;   // int4 in LDS

#if __has_builtin(__builtin_amdgcn_global_load_async_to_lds_b128)
__device__ __forceinline__ void async_b128(const void* g, void* l) {
  __builtin_amdgcn_global_load_async_to_lds_b128(
      (gv4p)(uintptr_t)g,                    // drops const, retypes to int4 AS1
      (lv4p)(unsigned)(uintptr_t)l,          // low 32 bits of generic == LDS offset
      0, 0);
}
#else
__device__ __forceinline__ void async_b128(const void* g, void* l) {
  unsigned lo = (unsigned)(uintptr_t)l;
  asm volatile("global_load_async_to_lds_b128 %0, %1, off"
               :: "v"(lo), "v"(g) : "memory");
}
#endif

__device__ __forceinline__ void wait_async0() {
#if __has_builtin(__builtin_amdgcn_s_wait_asynccnt)
  __builtin_amdgcn_s_wait_asynccnt(0);
#else
  asm volatile("s_wait_asynccnt 0" ::: "memory");
#endif
}

// ---------------------------------------------------------------------------
// Kernel 1: blocks [0,64)  : bias[b][o] = W1_b[o] + W2_b[o] + dot(h[b,:], W2_w[o,:])
//           blocks [64,128): convert W1 f32 -> bf16 into ws
// ---------------------------------------------------------------------------
__global__ void prep_kernel(const float* __restrict__ h,
                            const float* __restrict__ W1w,
                            const float* __restrict__ W1b,
                            const float* __restrict__ W2w,
                            const float* __restrict__ W2b,
                            float* __restrict__ biasw,
                            unsigned short* __restrict__ w1bf) {
  __shared__ float hs[HID];
  const int t = threadIdx.x;
  if (blockIdx.x < BATCH) {
    const int b = blockIdx.x;
    hs[t]       = h[b * HID + t];
    hs[t + 256] = h[b * HID + t + 256];
    __syncthreads();
    for (int j = t; j < HID; j += 256) {
      float acc = W1b[j] + W2b[j];
      const float* wr = W2w + j * HID;
#pragma unroll 4
      for (int k = 0; k < HID; ++k) acc = fmaf(hs[k], wr[k], acc);
      biasw[b * HID + j] = acc;
    }
  } else {
    const int base = (blockIdx.x - BATCH) * 4096;  // 64 blocks * 4096 = 512*512
#pragma unroll
    for (int i = 0; i < 16; ++i) {
      const int e = base + i * 256 + t;
      w1bf[e] = f2bf(W1w[e]);
    }
  }
}

// ---------------------------------------------------------------------------
// Kernel 2: 64-row tile GEMM (bf16 WMMA, f32 acc) fused with tanh + V-dot.
//   grid = 2048 blocks, 256 threads = 8 waves (4 M-groups x 2 N-halves).
//   A tile:   64 x 512 bf16 in LDS (64 KB), XOR-swizzled 16B chunks,
//             fully hoisted into 128 VGPRs of A-fragments per wave.
//   B chunks: async global->LDS double-buffered (2 x 32 KB), conflict-free
//             ds_load_b128 fragment reads; s_wait_asynccnt + barrier pipeline.
// ---------------------------------------------------------------------------
__global__ __launch_bounds__(256) void attn_gemm_kernel(
    const float* __restrict__ enc,
    const unsigned short* __restrict__ w1bf,
    const float* __restrict__ biasw,
    const float* __restrict__ Vw,
    float* __restrict__ out) {
  __shared__ unsigned short As[MTILE * HID];   // 64 KB
  __shared__ uint4 BsQ[2][2][16 * 64];         // 64 KB : [buf][N-half][col*64+chunk]
  const int t    = threadIdx.x;
  const int row0 = blockIdx.x * MTILE;         // flat row in [B*S, H]
  const int bidx = row0 >> 11;                 // row0 / SRC (tile never crosses batch)

  // Stage one 2x16-col, full-K B chunk (32 KB) with async b128 copies.
  auto stageB = [&](int nc, int buf) {
#pragma unroll
    for (int i = 0; i < 8; ++i) {
      const int e   = i * 256 + t;             // 2048 chunks of 16 B
      const int nh  = e >> 10;                 // N-half (which wn uses it)
      const int col = (e >> 6) & 15;
      const int ck  = e & 63;
      const int o   = nh * 256 + nc * 16 + col;
      async_b128(w1bf + o * HID + ck * 8,
                 &BsQ[buf][nh][col * 64 + (ck ^ col)]);
    }
  };

  stageB(0, 0);   // in flight while we convert/stage the A tile below

  // ---- Stage A tile: f32 -> bf16, 16B-chunk XOR swizzle (chunk ^ (row&15)) ----
  const float* encBase = enc + (size_t)row0 * HID;
  uint4* AsQ = (uint4*)As;
#pragma unroll
  for (int i = 0; i < 16; ++i) {
    const int e   = i * 256 + t;   // 4096 chunks of 8 bf16
    const int row = e >> 6;
    const int cid = e & 63;
    const float4* g = (const float4*)(encBase + row * HID + cid * 8);
    const float4 x0 = g[0], x1 = g[1];
    uint4 pk;
    pk.x = pack2bf(x0.x, x0.y); pk.y = pack2bf(x0.z, x0.w);
    pk.z = pack2bf(x1.x, x1.y); pk.w = pack2bf(x1.z, x1.w);
    AsQ[row * 64 + (cid ^ (row & 15))] = pk;
  }
  wait_async0();
  __syncthreads();

  const int lane = t & 31;
  const int wave = t >> 5;
  const int wm   = wave >> 1;          // 0..3 : which 16 rows
  const int wn   = wave & 1;           // 0..1 : which 256 output cols
  const int m0   = wm << 4;
  const int half = lane >> 4;          // A: K-offset {0,8}; B: K-offset {0,8}
  const int lrow = m0 + (lane & 15);
  const int rx   = lrow & 15;
  const int n    = lane & 15;

  // ---- Hoist all A fragments for K=512 (16 chunks of K=32) into registers ----
  const uint4* arow = AsQ + lrow * 64;
  ABFrag a[16];
#pragma unroll
  for (int kc = 0; kc < 16; ++kc) {
    const int c0 = 4 * kc + half;            // ISA A layout: lanes<16 K 0-7/16-23,
    a[kc].q[0] = arow[c0 ^ rx];              //               lanes>=16 K 8-15/24-31
    a[kc].q[1] = arow[(c0 + 2) ^ rx];
  }

  float acc[8] = {0.f, 0.f, 0.f, 0.f, 0.f, 0.f, 0.f, 0.f};
  const float* biasp = biasw + bidx * HID;

  for (int nc = 0; nc < 16; ++nc) {
    const int cur = nc & 1;
    if (nc + 1 < 16) stageB(nc + 1, cur ^ 1);   // overlap next stage with compute

    const uint4* brow = &BsQ[cur][wn][n * 64];
    v8f c = {0.f, 0.f, 0.f, 0.f, 0.f, 0.f, 0.f, 0.f};
#pragma unroll
    for (int kc = 0; kc < 16; ++kc) {
      ABFrag bb;                               // B: lanes<16 K 0-7/16-23, lanes>=16 K 8-15/24-31
      bb.q[0] = brow[(4 * kc + half)     ^ n];
      bb.q[1] = brow[(4 * kc + half + 2) ^ n];
      c = __builtin_amdgcn_wmma_f32_16x16x32_bf16(
              false, a[kc].v, false, bb.v, (short)0, c, false, false);
    }

    // Fused epilogue: score[m] += sum_n Vw[o] * tanh(c[m][n] + bias[b][o])
    const int o    = wn * 256 + nc * 16 + n;
    const float vw = Vw[o];
    const float bi = biasp[o];
#pragma unroll
    for (int r = 0; r < 8; ++r) {            // C layout: VGPR r -> M=r(+8), N=lane&15
      float e = fast_tanh(c[r] + bi) * vw;
      e += __shfl_xor(e, 1, 32);
      e += __shfl_xor(e, 2, 32);
      e += __shfl_xor(e, 4, 32);
      e += __shfl_xor(e, 8, 32);             // reduce the 16 N-lanes of each half
      acc[r] += e;
    }

    wait_async0();      // next-stage async writes landed
    __syncthreads();    // everyone done with buf[cur]; buf[cur^1] fully written
  }

  // ---- Combine the two N-halves via (reused) LDS, write raw scores ----
  float* sred = (float*)As;                  // loop ended on a barrier; As dead
  if ((lane & 15) == 0) {                    // lanes 0 (M=m0+r) and 16 (M=m0+8+r)
    const int base = wn * 64 + m0 + half * 8;
#pragma unroll
    for (int r = 0; r < 8; ++r) sred[base + r] = acc[r];
  }
  __syncthreads();
  if (t < MTILE) out[row0 + t] = sred[t] + sred[64 + t];  // V_b dropped (softmax-shift-invariant)
}

// ---------------------------------------------------------------------------
// Kernel 3: softmax over S per batch row, in place on d_out.
// ---------------------------------------------------------------------------
__global__ void softmax_kernel(float* __restrict__ out) {
  __shared__ float red[256];
  const int b = blockIdx.x, t = threadIdx.x;
  float* row = out + b * SRC;
  float v[8];
  float m = -INFINITY;
#pragma unroll
  for (int i = 0; i < 8; ++i) { v[i] = row[i * 256 + t]; m = fmaxf(m, v[i]); }
  red[t] = m; __syncthreads();
  for (int s = 128; s > 0; s >>= 1) {
    if (t < s) red[t] = fmaxf(red[t], red[t + s]);
    __syncthreads();
  }
  m = red[0]; __syncthreads();
  float sum = 0.f;
#pragma unroll
  for (int i = 0; i < 8; ++i) { v[i] = __expf(v[i] - m); sum += v[i]; }
  red[t] = sum; __syncthreads();
  for (int s = 128; s > 0; s >>= 1) {
    if (t < s) red[t] += red[t + s];
    __syncthreads();
  }
  const float inv = 1.f / red[0];
#pragma unroll
  for (int i = 0; i < 8; ++i) row[i * 256 + t] = v[i] * inv;
}

// ---------------------------------------------------------------------------
extern "C" void kernel_launch(void* const* d_in, const int* in_sizes, int n_in,
                              void* d_out, int out_size, void* d_ws, size_t ws_size,
                              hipStream_t stream) {
  const float* h   = (const float*)d_in[0];
  const float* enc = (const float*)d_in[1];
  const float* W1w = (const float*)d_in[2];
  const float* W1b = (const float*)d_in[3];
  const float* W2w = (const float*)d_in[4];
  const float* W2b = (const float*)d_in[5];
  const float* Vw  = (const float*)d_in[6];
  // d_in[7] = V_b: uniform additive shift, cancelled by softmax -> unused.
  float* out = (float*)d_out;

  char* wsb = (char*)d_ws;
  float*          biasw = (float*)wsb;                                   // 128 KB
  unsigned short* w1bf  = (unsigned short*)(wsb + BATCH * HID * 4);      // 512 KB

  prep_kernel<<<128, 256, 0, stream>>>(h, W1w, W1b, W2w, W2b, biasw, w1bf);
  attn_gemm_kernel<<<(BATCH * SRC) / MTILE, 256, 0, stream>>>(enc, w1bf, biasw, Vw, out);
  softmax_kernel<<<BATCH, 256, 0, stream>>>(out);
}
